// ADTM_73237782331696
// MI455X (gfx1250) — compile-verified, matrix operand order checked by reading, not run on hardware
//
#include <hip/hip_runtime.h>
#include <hip/hip_bf16.h>
#include <math.h>

typedef __attribute__((ext_vector_type(16))) _Float16 v16h;
typedef __attribute__((ext_vector_type(8)))  _Float16 h8;
typedef __attribute__((ext_vector_type(8)))  float    v8f;
typedef __attribute__((ext_vector_type(4)))  unsigned int v4u;
typedef __attribute__((ext_vector_type(8)))  int      v8i;
typedef __attribute__((ext_vector_type(4)))  int      v4i;

#define HTOT 112
#define WTOT 112
#define CH   64
#define TAPS 25
#define PIX_PER_IMG (HTOT*WTOT)            // 12544
#define NPIX (2*PIX_PER_IMG)               // 25088
#define NTILES (NPIX/16)                   // 1568
#define WAVES_PER_BLOCK 8
#define GEMM_BLOCKS (NTILES/WAVES_PER_BLOCK) // 196
#define FRAG_HALFS (TAPS*2*4*32*16)        // 102400 f16 elems
#define FRAG_BYTES (FRAG_HALFS*2)          // 204800 B  (fits in 320KB WGP LDS)

// ---------------- helpers ----------------
static __device__ inline h8 splat8(_Float16 s) {
    h8 r = { s, s, s, s, s, s, s, s };
    return r;
}

// Stage the whole B-fragment table (204,800 B) into LDS.
// Preferred path: Tensor Data Mover (one DMA descriptor, wave 0 issues,
// waits TENSORcnt, block barrier). Fallback: cooperative uint4 copy.
static __device__ inline void stage_frags_to_lds(const _Float16* __restrict__ gsrc,
                                                 _Float16* sdst) {
#if __has_builtin(__builtin_amdgcn_tensor_load_to_lds)
    if (threadIdx.x < 32) {                      // whole wave 0, EXEC ignored by TDM
        unsigned long long ga = (unsigned long long)gsrc;
        unsigned lds_off = (unsigned)(unsigned long long)sdst; // low 32b = LDS offset
        const unsigned n8 = FRAG_BYTES / 8;      // 25600 8-byte elements (1-D tile)
        // D# group 0: count=1 | lds_addr | global_addr[56:0] | type=2
        v4u g0 = { 1u,
                   lds_off,
                   (unsigned)ga,
                   (unsigned)((ga >> 32) & 0x1FFFFFFu) | (2u << 30) };
        // D# group 1: data_size=8B; tensor_dim0=tile_dim0=n8; dim1=1; strides=n8
        v8i g1 = { (int)(3u << 16),                                   // data_size=8B
                   (int)((n8 & 0xFFFFu) << 16),                       // dim0[15:0]
                   (int)(((n8 >> 16) & 0xFFFFu) | (1u << 16)),        // dim0[31:16] | dim1 lo=1
                   (int)((n8 & 0xFFFFu) << 16),                       // tile_dim0
                   1,                                                 // tile_dim1=1
                   (int)n8,                                           // dim0_stride[31:0]
                   (int)(((n8 >> 16) & 0xFFFFu) | ((n8 & 0xFFFFu) << 16)), // stride0 hi | stride1 lo
                   0 };
        v4i z4 = { 0, 0, 0, 0 };
#if __clang_major__ >= 23
        v8i z8 = { 0, 0, 0, 0, 0, 0, 0, 0 };
        __builtin_amdgcn_tensor_load_to_lds(g0, g1, z4, z4, z8, 0);
#else
        __builtin_amdgcn_tensor_load_to_lds(g0, g1, z4, z4, 0);
#endif
        __builtin_amdgcn_s_wait_tensorcnt(0);
    }
#else
    const uint4* gs = (const uint4*)gsrc;
    uint4* ds = (uint4*)sdst;
    for (int i = threadIdx.x; i < (int)(FRAG_BYTES / 16); i += 256) ds[i] = gs[i];
#endif
    __syncthreads();
}

// ---------------- kernel 0: NCHW f32 -> NHWC f16 ----------------
__global__ __launch_bounds__(256) void k_convert(const float* __restrict__ x,
                                                 _Float16* __restrict__ xt) {
    int i = blockIdx.x * 256 + threadIdx.x;           // over B*H*W*C (c innermost)
    if (i >= NPIX * CH) return;
    int c  = i & (CH - 1);
    int p  = i >> 6;
    int b  = p / PIX_PER_IMG;
    int yx = p - b * PIX_PER_IMG;
    xt[i] = (_Float16)x[(b * CH + c) * PIX_PER_IMG + yx];
}

// ---------------- kernel 1: pack weights into WMMA B fragments ----------------
// frag layout: [(t*2+ks)*4+nt][lane 0..31][16 f16]
// B(32x16 f16) per-lane: n = lane&15; kb = (lane&16)?8:0;
//   elem i<8  -> K = kb + i ;  elem i>=8 -> K = kb + 16 + (i-8)
__global__ __launch_bounds__(256) void k_pack(const float* __restrict__ w,  // [O][64][25]
                                              int o_valid,
                                              _Float16* __restrict__ frag) {
    int tid = blockIdx.x * 256 + threadIdx.x;         // 25*2*4*32 = 6400
    if (tid >= TAPS * 2 * 4 * 32) return;
    int lane = tid & 31;
    int rest = tid >> 5;
    int nt = rest & 3; rest >>= 2;
    int ks = rest & 1;
    int t  = rest >> 1;
    int n  = nt * 16 + (lane & 15);
    int kb = (lane & 16) ? 8 : 0;
    _Float16* dst = frag + (size_t)tid * 16;
    #pragma unroll
    for (int i = 0; i < 16; i++) {
        int kloc = kb + ((i < 8) ? i : (8 + i));      // 0..31 within K-step
        int c = ks * 32 + kloc;
        float v = (n < o_valid) ? w[((size_t)n * CH + c) * TAPS + t] : 0.0f;
        dst[i] = (_Float16)v;
    }
}

// ---------------- kernel 2: offset conv (implicit GEMM, WMMA, B in LDS) ----------------
__global__ __launch_bounds__(256) void k_offconv(const _Float16* __restrict__ xt,
                                                 const _Float16* __restrict__ wfrag,
                                                 const float* __restrict__ b_off,
                                                 float* __restrict__ off_cl) {
    extern __shared__ __align__(16) _Float16 sfrag[];
    stage_frags_to_lds(wfrag, sfrag);

    const int lane = threadIdx.x & 31;
    const int wave = threadIdx.x >> 5;
    const int tile = blockIdx.x * WAVES_PER_BLOCK + wave;
    const int half = lane >> 4;
    const int m    = lane & 15;
    const int p    = tile * 16 + m;
    const int b    = p / PIX_PER_IMG;
    const int yx   = p - b * PIX_PER_IMG;
    const int y    = yx / WTOT;
    const int xx   = yx - y * WTOT;
    const int ch0  = half * 8;

    v8f acc[4] = {v8f{}, v8f{}, v8f{}, v8f{}};

    #pragma unroll 1
    for (int t = 0; t < TAPS; t++) {
        int ky = t / 5, kx = t - 5 * (t / 5);
        int sy = y + ky - 2, sx = xx + kx - 2;
        bool valid = (sy >= 0) & (sy < HTOT) & (sx >= 0) & (sx < WTOT);
        const _Float16* src = xt + (((size_t)(b * HTOT + sy) * WTOT + sx) << 6);
        #pragma unroll
        for (int ks = 0; ks < 2; ks++) {
            h8 lo = {}, hi = {};
            if (valid) {
                lo = *(const h8*)(src + ks * 32 + ch0);
                hi = *(const h8*)(src + ks * 32 + ch0 + 16);
            }
            v16h a;
            #pragma unroll
            for (int i = 0; i < 8; i++) { a[i] = lo[i]; a[i + 8] = hi[i]; }
            const _Float16* fb = sfrag + ((size_t)((t * 2 + ks) * 4) * 32 + lane) * 16;
            #pragma unroll
            for (int nt = 0; nt < 4; nt++) {
                v16h bm = *(const v16h*)(fb + (size_t)nt * 32 * 16);
                acc[nt] = __builtin_amdgcn_wmma_f32_16x16x32_f16(
                    false, a, false, bm, (short)0, acc[nt], false, false);
            }
        }
    }

    const int n0 = lane & 15;
    #pragma unroll
    for (int nt = 0; nt < 4; nt++) {
        int o = nt * 16 + n0;
        if (o < 2 * TAPS) {
            float bias = b_off[o];
            #pragma unroll
            for (int r = 0; r < 8; r++) {
                int p2 = tile * 16 + r + half * 8;
                off_cl[(size_t)p2 * (2 * TAPS) + o] = acc[nt][r] + bias;
            }
        }
    }
}

// ---------------- kernel 3: fused bilinear deform-sample + main GEMM (B in LDS) ----------------
__global__ __launch_bounds__(256) void k_deform(const _Float16* __restrict__ xt,
                                                const _Float16* __restrict__ wfrag,
                                                const float* __restrict__ off_cl,
                                                const float* __restrict__ b_def,
                                                float* __restrict__ out) {
    extern __shared__ __align__(16) _Float16 sfrag[];
    stage_frags_to_lds(wfrag, sfrag);

    const int lane = threadIdx.x & 31;
    const int wave = threadIdx.x >> 5;
    const int tile = blockIdx.x * WAVES_PER_BLOCK + wave;
    const int half = lane >> 4;
    const int m    = lane & 15;
    const int p    = tile * 16 + m;
    const int b    = p / PIX_PER_IMG;
    const int yx   = p - b * PIX_PER_IMG;
    const int y    = yx / WTOT;
    const int xx   = yx - y * WTOT;
    const int ch0  = half * 8;

    v8f acc[4] = {v8f{}, v8f{}, v8f{}, v8f{}};

    #pragma unroll 1
    for (int t = 0; t < TAPS; t++) {
        int ky = t / 5, kx = t - 5 * (t / 5);
        float dy = off_cl[(size_t)p * (2 * TAPS) + 2 * t];
        float dx = off_cl[(size_t)p * (2 * TAPS) + 2 * t + 1];
        float py = (float)(y + ky - 2) + dy;
        float px = (float)(xx + kx - 2) + dx;
        float fy0 = floorf(py), fx0 = floorf(px);
        int y0 = (int)fy0, x0 = (int)fx0;
        float wy1 = py - fy0, wx1 = px - fx0;
        _Float16 h00 = (_Float16)((1.0f - wy1) * (1.0f - wx1));
        _Float16 h01 = (_Float16)((1.0f - wy1) * wx1);
        _Float16 h10 = (_Float16)(wy1 * (1.0f - wx1));
        _Float16 h11 = (_Float16)(wy1 * wx1);
        bool vy0 = (y0 >= 0) & (y0 < HTOT);
        bool vy1 = (y0 + 1 >= 0) & (y0 + 1 < HTOT);
        bool vx0 = (x0 >= 0) & (x0 < WTOT);
        bool vx1 = (x0 + 1 >= 0) & (x0 + 1 < WTOT);
        const _Float16* base = xt + ((size_t)b * PIX_PER_IMG << 6);
        const _Float16* p00 = base + (((size_t)y0 * WTOT + x0) << 6);
        const _Float16* p01 = p00 + 64;
        const _Float16* p10 = p00 + ((size_t)WTOT << 6);
        const _Float16* p11 = p10 + 64;

        #pragma unroll
        for (int ks = 0; ks < 2; ks++) {
            int cb = ks * 32 + ch0;
            h8 lo = {}, hi = {};
            if (vy0 & vx0) { lo += (*(const h8*)(p00 + cb)) * splat8(h00);
                             hi += (*(const h8*)(p00 + cb + 16)) * splat8(h00); }
            if (vy0 & vx1) { lo += (*(const h8*)(p01 + cb)) * splat8(h01);
                             hi += (*(const h8*)(p01 + cb + 16)) * splat8(h01); }
            if (vy1 & vx0) { lo += (*(const h8*)(p10 + cb)) * splat8(h10);
                             hi += (*(const h8*)(p10 + cb + 16)) * splat8(h10); }
            if (vy1 & vx1) { lo += (*(const h8*)(p11 + cb)) * splat8(h11);
                             hi += (*(const h8*)(p11 + cb + 16)) * splat8(h11); }
            v16h a;
            #pragma unroll
            for (int i = 0; i < 8; i++) { a[i] = lo[i]; a[i + 8] = hi[i]; }
            const _Float16* fb = sfrag + ((size_t)((t * 2 + ks) * 4) * 32 + lane) * 16;
            #pragma unroll
            for (int nt = 0; nt < 4; nt++) {
                v16h bm = *(const v16h*)(fb + (size_t)nt * 32 * 16);
                acc[nt] = __builtin_amdgcn_wmma_f32_16x16x32_f16(
                    false, a, false, bm, (short)0, acc[nt], false, false);
            }
        }
    }

    const int n0 = lane & 15;
    #pragma unroll
    for (int nt = 0; nt < 4; nt++) {
        int o = nt * 16 + n0;
        float bias = b_def[o];
        #pragma unroll
        for (int r = 0; r < 8; r++) {
            int p2 = tile * 16 + r + half * 8;
            int b2 = p2 / PIX_PER_IMG;
            int yx2 = p2 - b2 * PIX_PER_IMG;
            out[((size_t)(b2 * CH + o)) * PIX_PER_IMG + yx2] = acc[nt][r] + bias;
        }
    }
}

// ---------------- kernel 4: GAP of |out| per (b, c) ----------------
__global__ __launch_bounds__(256) void k_gap(const float* __restrict__ out,
                                             float* __restrict__ gap) {
    __shared__ float red[256];
    int bc = blockIdx.x;                              // 0..127
    const float* src = out + (size_t)bc * PIX_PER_IMG;
    float s = 0.0f;
    for (int i = threadIdx.x; i < PIX_PER_IMG; i += 256) s += fabsf(src[i]);
    red[threadIdx.x] = s;
    __syncthreads();
    for (int off = 128; off > 0; off >>= 1) {
        if (threadIdx.x < off) red[threadIdx.x] += red[threadIdx.x + off];
        __syncthreads();
    }
    if (threadIdx.x == 0) gap[bc] = red[0] * (1.0f / (float)PIX_PER_IMG);
}

// ---------------- kernel 5: two 64x64 FCs -> per-channel threshold tau ----------------
__global__ __launch_bounds__(128) void k_fc(const float* __restrict__ gap,
                                            const float* __restrict__ w1,
                                            const float* __restrict__ b1,
                                            const float* __restrict__ w2,
                                            const float* __restrict__ b2,
                                            float* __restrict__ tau) {
    __shared__ float z[2][64];
    int tid = threadIdx.x;                            // 0..127
    int bb = tid >> 6, c = tid & 63;
    float s = b1[c];
    for (int k = 0; k < 64; k++) s += gap[bb * 64 + k] * w1[k * 64 + c];
    z[bb][c] = fmaxf(s, 0.0f);
    __syncthreads();
    float s2 = b2[c];
    for (int k = 0; k < 64; k++) s2 += z[bb][k] * w2[k * 64 + c];
    float alpha = 1.0f / (1.0f + expf(-s2));
    tau[bb * 64 + c] = alpha * gap[bb * 64 + c];
}

// ---------------- kernel 6: in-place soft threshold ----------------
__global__ __launch_bounds__(256) void k_shrink(float* __restrict__ out,
                                                const float* __restrict__ tau) {
    int i = blockIdx.x * 256 + threadIdx.x;
    if (i >= NPIX * CH) return;
    int bc = i / PIX_PER_IMG;                         // b*64 + c
    float v = out[i];
    float a = fabsf(v) - tau[bc];
    out[i] = (a > 0.0f) ? copysignf(a, v) : 0.0f;
}

// ---------------- launcher ----------------
extern "C" void kernel_launch(void* const* d_in, const int* in_sizes, int n_in,
                              void* d_out, int out_size, void* d_ws, size_t ws_size,
                              hipStream_t stream) {
    const float* x     = (const float*)d_in[0];
    const float* w_off = (const float*)d_in[1];
    const float* b_off = (const float*)d_in[2];
    const float* w_def = (const float*)d_in[3];
    const float* b_def = (const float*)d_in[4];
    const float* w1    = (const float*)d_in[5];
    const float* b1    = (const float*)d_in[6];
    const float* w2    = (const float*)d_in[7];
    const float* b2    = (const float*)d_in[8];
    float* out = (float*)d_out;

    char* ws = (char*)d_ws;
    const size_t XT_BYTES   = (size_t)NPIX * CH * sizeof(_Float16);      // 3,211,264
    const size_t OFF_BYTES  = (size_t)NPIX * (2 * TAPS) * sizeof(float); // 5,017,600
    _Float16* xt        = (_Float16*)ws;                 ws += XT_BYTES;
    _Float16* frag_off  = (_Float16*)ws;                 ws += FRAG_BYTES;
    _Float16* frag_def  = (_Float16*)ws;                 ws += FRAG_BYTES;
    float*    off_cl    = (float*)ws;                    ws += OFF_BYTES;
    float*    gap       = (float*)ws;                    ws += 512;
    float*    tau       = (float*)ws;

    const int NEL = NPIX * CH;
    k_convert<<<(NEL + 255) / 256, 256, 0, stream>>>(x, xt);
    k_pack<<<(TAPS * 2 * 4 * 32 + 255) / 256, 256, 0, stream>>>(w_off, 2 * TAPS, frag_off);
    k_pack<<<(TAPS * 2 * 4 * 32 + 255) / 256, 256, 0, stream>>>(w_def, CH, frag_def);
    k_offconv<<<GEMM_BLOCKS, 256, FRAG_BYTES, stream>>>(xt, frag_off, b_off, off_cl);
    k_deform<<<GEMM_BLOCKS, 256, FRAG_BYTES, stream>>>(xt, frag_def, off_cl, b_def, out);
    k_gap<<<2 * CH, 256, 0, stream>>>(out, gap);
    k_fc<<<1, 128, 0, stream>>>(gap, w1, b1, w2, b2, tau);
    k_shrink<<<(NEL + 255) / 256, 256, 0, stream>>>(out, tau);
}
